// SigLipLoss_84834194031233
// MI455X (gfx1250) — matrix operand order repaired and checked by reading, not compiled, not run on hardware
//
#include <hip/hip_runtime.h>
#include <hip/hip_bf16.h>

// ---------------------------------------------------------------------------
// SigLIP loss, fused: logits tile (bf16 WMMA, f32 accum) -> log-sigmoid ->
// block partial sums -> deterministic final reduction.  Never materializes
// the 16384^2 logit matrix (that would be ~2 GB of HBM traffic; the whole
// 33.5 MB input set is L2-resident on MI455X's 192 MB L2).
// Epilogue uses hardware transcendentals (v_exp_f32 / v_log_f32) instead of
// the log1pf libcall to keep the static code size dominated by the WMMA loop.
// ---------------------------------------------------------------------------

typedef __bf16 bf16_t;
typedef bf16_t v16bf __attribute__((ext_vector_type(16)));
typedef bf16_t v8bf  __attribute__((ext_vector_type(8)));
typedef bf16_t v4bf  __attribute__((ext_vector_type(4)));
typedef float  v8f   __attribute__((ext_vector_type(8)));
typedef float  v4f   __attribute__((ext_vector_type(4)));

#define DDIM 256          // embedding dim (K)
#define BM   128          // tile rows  (image side)
#define BN   128          // tile cols  (text side)
#define KPAD 8            // pad row stride: 264 elems -> row shift = 4 dwords mod 64 banks
#define LDK  (DDIM + KPAD)

// A fragment (16x32 bf16, MxK): lanes 0-15 = rows M=0..15 holding K[0..7],K[16..23];
// lanes 16-31 = same rows holding K[8..15],K[24..31]   (ISA 7.12.2)
__device__ __forceinline__ v16bf load_frag_a(const bf16_t* __restrict__ base, int lane) {
    const int r    = lane & 15;
    const int koff = (lane >> 4) << 3;               // 0 or 8
    const bf16_t* p = base + r * LDK + koff;
    v8bf lo = *(const v8bf*)(p);
    v8bf hi = *(const v8bf*)(p + 16);
    return __builtin_shufflevector(lo, hi, 0,1,2,3,4,5,6,7,8,9,10,11,12,13,14,15);
}

// B fragment (32x16 bf16, KxN): lane n holds column n, lanes 0-15 K[0..7],K[16..23],
// lanes 16-31 K[8..15],K[24..31] (matches A).  B = txt^T and txt is row-major, so a
// "column" of B is a contiguous row of the text tile — no transpose needed.
__device__ __forceinline__ v16bf load_frag_b(const bf16_t* __restrict__ base, int lane) {
    const int c    = lane & 15;
    const int koff = (lane >> 4) << 3;               // 0 or 8
    const bf16_t* p = base + c * LDK + koff;
    v8bf lo = *(const v8bf*)(p);
    v8bf hi = *(const v8bf*)(p + 16);
    return __builtin_shufflevector(lo, hi, 0,1,2,3,4,5,6,7,8,9,10,11,12,13,14,15);
}

extern "C" __global__ __launch_bounds__(256)
void siglip_tile_kernel(const float* __restrict__ img,
                        const float* __restrict__ txt,
                        const float* __restrict__ scale_p,
                        const float* __restrict__ bias_p,
                        float* __restrict__ partial) {
    extern __shared__ char smem_raw[];
    bf16_t* As  = (bf16_t*)smem_raw;                 // [BM][LDK]
    bf16_t* Bs  = As + (size_t)BM * LDK;             // [BN][LDK]
    float*  red = (float*)(Bs + (size_t)BN * LDK);   // [256]

    const int  tid   = threadIdx.x;
    const int  lane  = tid & 31;
    const int  wave  = tid >> 5;                     // 8 waves
    const int  wm    = wave & 3;                     // M offset 32*wm
    const int  wn    = wave >> 2;                    // N offset 64*wn
    const long tileM = (long)blockIdx.y * BM;
    const long tileN = (long)blockIdx.x * BN;

    // ---- stage full-K tiles: global f32 -> cvt bf16 -> LDS (padded rows) ----
    for (int i = tid; i < BM * (DDIM / 4); i += 256) {
        const int row = i >> 6;                      // 64 float4 quads per row
        const int q   = (i & 63) << 2;
        v4f va = *(const v4f*)(img + (tileM + row) * DDIM + q);
        v4f vb = *(const v4f*)(txt + (tileN + row) * DDIM + q);
        v4bf ba = { (bf16_t)va.x, (bf16_t)va.y, (bf16_t)va.z, (bf16_t)va.w };
        v4bf bb = { (bf16_t)vb.x, (bf16_t)vb.y, (bf16_t)vb.z, (bf16_t)vb.w };
        *(v4bf*)(As + row * LDK + q) = ba;
        *(v4bf*)(Bs + row * LDK + q) = bb;
    }
    __syncthreads();

    // ---- 32x64 per wave: 2x4 accum tiles, K fully unrolled (64 WMMAs/wave) ----
    v8f acc[2][4];
    #pragma unroll
    for (int mi = 0; mi < 2; ++mi)
        #pragma unroll
        for (int ni = 0; ni < 4; ++ni)
            #pragma unroll
            for (int e = 0; e < 8; ++e) acc[mi][ni][e] = 0.0f;

    #pragma unroll
    for (int k0 = 0; k0 < DDIM; k0 += 32) {
        v16bf a0 = load_frag_a(As + (wm * 32 +  0) * LDK + k0, lane);
        v16bf a1 = load_frag_a(As + (wm * 32 + 16) * LDK + k0, lane);
        #pragma unroll
        for (int ni = 0; ni < 4; ++ni) {
            v16bf b = load_frag_b(Bs + (wn * 64 + ni * 16) * LDK + k0, lane);
            acc[0][ni] = __builtin_amdgcn_wmma_f32_16x16x32_bf16(
                false, a0, false, b, (short)0, acc[0][ni], false, false);
            acc[1][ni] = __builtin_amdgcn_wmma_f32_16x16x32_bf16(
                false, a1, false, b, (short)0, acc[1][ni], false, false);
        }
    }

    // ---- fused epilogue: logits -> labels*logits -> softplus, per-thread sum ----
    // C/D layout: VGPR r, lanes 0-15 -> (M=r, N=lane); lanes 16-31 -> (M=r+8, N=lane-16)
    // -log_sigmoid(s) = softplus(-s) = max(-s,0) + log(1 + exp(-|s|)); with
    // t = -s this is max(t,0) + log(1 + exp(-|t|)).  exp argument <= 0, so the
    // log argument is in (1,2] and hardware __logf is safe and accurate here.
    const float scale = *scale_p;
    const float bias  = *bias_p;
    const int   half  = lane >> 4;
    const int   col   = lane & 15;
    float lsum = 0.0f;
    #pragma unroll
    for (int mi = 0; mi < 2; ++mi) {
        #pragma unroll
        for (int ni = 0; ni < 4; ++ni) {
            const long gN  = tileN + wn * 64 + ni * 16 + col;
            const long gM0 = tileM + wm * 32 + mi * 16 + half * 8;
            #pragma unroll
            for (int r = 0; r < 8; ++r) {
                const float logit = fmaf(scale, acc[mi][ni][r], bias);
                // t = -signed = -(labels*logit): diagonal -> -logit, else +logit
                const float t = ((gM0 + r) == gN) ? -logit : logit;
                lsum += fmaxf(t, 0.0f) + __logf(1.0f + __expf(-fabsf(t)));
            }
        }
    }

    // ---- deterministic fixed-pairing block reduction ----
    red[tid] = lsum;
    __syncthreads();
    for (int s = 128; s > 0; s >>= 1) {
        if (tid < s) red[tid] += red[tid + s];
        __syncthreads();
    }
    if (tid == 0) partial[(long)blockIdx.y * gridDim.x + blockIdx.x] = red[0];
}

extern "C" __global__ __launch_bounds__(256)
void siglip_reduce_kernel(const float* __restrict__ partial, int n_part,
                          float inv_n, float* __restrict__ out) {
    __shared__ float red[256];
    float s = 0.0f;
    for (int i = threadIdx.x; i < n_part; i += 256) s += partial[i];  // fixed order
    red[threadIdx.x] = s;
    __syncthreads();
    for (int t = 128; t > 0; t >>= 1) {
        if (threadIdx.x < t) red[threadIdx.x] += red[threadIdx.x + t];
        __syncthreads();
    }
    if (threadIdx.x == 0) out[0] = red[0] * inv_n;   // loss = sum(softplus(-signed))/N
}

extern "C" void kernel_launch(void* const* d_in, const int* in_sizes, int n_in,
                              void* d_out, int out_size, void* d_ws, size_t ws_size,
                              hipStream_t stream) {
    const float* img     = (const float*)d_in[0];
    const float* txt     = (const float*)d_in[1];
    const float* scale_p = (const float*)d_in[2];
    const float* bias_p  = (const float*)d_in[3];
    // d_in[4] (curvature) unused for geometry='clip'

    const int n     = in_sizes[0] / DDIM;            // 16384
    const int tiles = n / BM;                        // 128
    float* partial  = (float*)d_ws;                  // tiles*tiles floats = 64 KB

    const size_t shmem = (size_t)(BM + BN) * LDK * sizeof(bf16_t) + 256 * sizeof(float);
    (void)hipFuncSetAttribute((const void*)siglip_tile_kernel,
                              hipFuncAttributeMaxDynamicSharedMemorySize, (int)shmem);

    dim3 grid(tiles, tiles);
    siglip_tile_kernel<<<grid, 256, shmem, stream>>>(img, txt, scale_p, bias_p, partial);
    siglip_reduce_kernel<<<1, 256, 0, stream>>>(partial, tiles * tiles,
                                                1.0f / (float)n, (float*)d_out);
}
